// Expert_44538810860342
// MI455X (gfx1250) — matrix operand order repaired and checked by reading, not compiled, not run on hardware
//
#include <hip/hip_runtime.h>
#include <hip/hip_bf16.h>

typedef _Float16 half_t;
typedef __attribute__((ext_vector_type(16))) _Float16 v16h;
typedef __attribute__((ext_vector_type(8)))  _Float16 v8h;
typedef __attribute__((ext_vector_type(8)))  float    v8f;

#define NEXPERT 8

// ---------------------------------------------------------------------------
// WMMA helper: D = A(16x32 f16) x B(32x16 f16) + C(16x16 f32)
// ---------------------------------------------------------------------------
__device__ __forceinline__ v8f wmma_f16(v16h a, v16h b, v8f c) {
  return __builtin_amdgcn_wmma_f32_16x16x32_f16(
      /*neg_a=*/false, a, /*neg_b=*/false, b,
      /*c_mod=*/(short)0, c, /*reuse_a=*/false, /*reuse_b=*/false);
}

// A-fragment (16x32, M=rows): lane l holds row (l&15); VGPR0-3 = K at
// base+8*(l>>4) .. +7, VGPR4-7 = K at base+8*(l>>4)+16 .. +23.
// Caller passes p already offset by row*K + k0 + 8*(lane>>4).
__device__ __forceinline__ v16h load_a_frag(const half_t* p) {
  v16h r;
  ((v8h*)&r)[0] = *(const v8h*)(p);
  ((v8h*)&r)[1] = *(const v8h*)(p + 16);
  return r;
}

// ---------------------------------------------------------------------------
// fp32 -> (hi,lo) f16 split: x ~= hi + lo with ~21 effective mantissa bits.
// ---------------------------------------------------------------------------
__global__ void split_kernel(const float* __restrict__ src,
                             half_t* __restrict__ hi, half_t* __restrict__ lo,
                             int n) {
  int i = blockIdx.x * blockDim.x + threadIdx.x;
  if (i < n) {
    float v = src[i];
    half_t h = (half_t)v;
    hi[i] = h;
    lo[i] = (half_t)(v - (float)h);
  }
}

template<int O, bool ACT, bool OUTF>
__device__ __forceinline__ void emit_one(float y, int b, int o,
                                         half_t* __restrict__ oHi,
                                         half_t* __restrict__ oLo,
                                         float* __restrict__ oF) {
  if (ACT) y = (y > 0.0f) ? y : (__expf(y) - 1.0f);   // ELU, alpha = 1
  size_t idx = (size_t)b * O + o;
  if (OUTF) {
    oF[idx] = y;
  } else {
    half_t h = (half_t)y;
    oHi[idx] = h;
    oLo[idx] = (half_t)(y - (float)h);
  }
}

// ---------------------------------------------------------------------------
// One blended layer: out[b,o] = act( sum_e blend[b,e] * ((a @ W_e^T)[b,o]
//                                    + bias[e,o]) )
// Block: 128 threads = 4 waves. Block tile 32(batch) x 128(out).
// Each wave: 32x32 region = 2x2 WMMA tiles, split-f16 (3 WMMAs per tile).
// ---------------------------------------------------------------------------
template<int K, int O, bool ACT, bool OUTF>
__global__ __launch_bounds__(128) void layer_kernel(
    const half_t* __restrict__ aHi, const half_t* __restrict__ aLo,
    const half_t* __restrict__ wHi, const half_t* __restrict__ wLo,
    const float*  __restrict__ blend, const float* __restrict__ bias,
    half_t* __restrict__ oHi, half_t* __restrict__ oLo,
    float*  __restrict__ oF) {
  const int lane  = threadIdx.x & 31;
  const int wave  = threadIdx.x >> 5;
  const int lm    = lane & 15;
  const int lh    = lane >> 4;
  const int bBase = blockIdx.y * 32;
  const int oBase = blockIdx.x * 128 + wave * 32;

  const int col0 = oBase + lm;
  const int col1 = col0 + 16;
  const size_t ar0 = (size_t)(bBase + lm) * K;        // A row, m-tile 0
  const size_t ar1 = (size_t)(bBase + 16 + lm) * K;   // A row, m-tile 1

  v8f f00 = {}, f01 = {}, f10 = {}, f11 = {};         // final f32 accumulators

#pragma unroll 1
  for (int e = 0; e < NEXPERT; ++e) {
    const half_t* wbHi = wHi + (size_t)e * O * K;
    const half_t* wbLo = wLo + (size_t)e * O * K;
    v8f e00 = {}, e01 = {}, e10 = {}, e11 = {};       // per-expert accumulators

#pragma unroll 2
    for (int k0 = 0; k0 < K; k0 += 32) {
      const int ka = k0 + lh * 8;    // A-fragment K base for this lane half
      const int kb = k0 + lh * 8;    // B-fragment K base (same split as A)

      v16h a0h = load_a_frag(aHi + ar0 + ka);
      v16h a0l = load_a_frag(aLo + ar0 + ka);
      v16h a1h = load_a_frag(aHi + ar1 + ka);
      v16h a1l = load_a_frag(aLo + ar1 + ka);

      v16h b0h = load_a_frag(wbHi + (size_t)col0 * K + kb);
      v16h b0l = load_a_frag(wbLo + (size_t)col0 * K + kb);
      v16h b1h = load_a_frag(wbHi + (size_t)col1 * K + kb);
      v16h b1l = load_a_frag(wbLo + (size_t)col1 * K + kb);

      // hi*hi
      e00 = wmma_f16(a0h, b0h, e00);
      e01 = wmma_f16(a0h, b1h, e01);
      e10 = wmma_f16(a1h, b0h, e10);
      e11 = wmma_f16(a1h, b1h, e11);
      // hi*lo
      e00 = wmma_f16(a0h, b0l, e00);
      e01 = wmma_f16(a0h, b1l, e01);
      e10 = wmma_f16(a1h, b0l, e10);
      e11 = wmma_f16(a1h, b1l, e11);
      // lo*hi  (lo*lo term ~2^-22 relative: dropped)
      e00 = wmma_f16(a0l, b0h, e00);
      e01 = wmma_f16(a0l, b1h, e01);
      e10 = wmma_f16(a1l, b0h, e10);
      e11 = wmma_f16(a1l, b1h, e11);
    }

    // Blend combine in exact f32; bias folds in: blend*(acc + bias_e[o]).
    // C/D layout: VGPR r, lane l -> M = r + 8*(l>>4), N = l&15.
    const float bb0 = bias[e * O + col0];
    const float bb1 = bias[e * O + col1];
    const int rb0 = bBase + lh * 8;
#pragma unroll
    for (int r = 0; r < 8; ++r) {
      float bl0 = blend[(size_t)(rb0 + r) * NEXPERT + e];
      float bl1 = blend[(size_t)(rb0 + 16 + r) * NEXPERT + e];
      f00[r] += bl0 * (e00[r] + bb0);
      f01[r] += bl0 * (e01[r] + bb1);
      f10[r] += bl1 * (e10[r] + bb0);
      f11[r] += bl1 * (e11[r] + bb1);
    }
  }

  const int rb = bBase + lh * 8;
#pragma unroll
  for (int r = 0; r < 8; ++r) {
    const int b0 = rb + r;
    const int b1 = b0 + 16;
    emit_one<O, ACT, OUTF>(f00[r], b0, col0, oHi, oLo, oF);
    emit_one<O, ACT, OUTF>(f01[r], b0, col1, oHi, oLo, oF);
    emit_one<O, ACT, OUTF>(f10[r], b1, col0, oHi, oLo, oF);
    emit_one<O, ACT, OUTF>(f11[r], b1, col1, oHi, oLo, oF);
  }
}

// ---------------------------------------------------------------------------
extern "C" void kernel_launch(void* const* d_in, const int* in_sizes, int n_in,
                              void* d_out, int out_size, void* d_ws, size_t ws_size,
                              hipStream_t stream) {
  const float* blend = (const float*)d_in[0];
  const float* x     = (const float*)d_in[1];
  const float* W0    = (const float*)d_in[2];
  const float* B0    = (const float*)d_in[3];
  const float* W1    = (const float*)d_in[4];
  const float* B1    = (const float*)d_in[5];
  const float* W2    = (const float*)d_in[6];
  const float* B2    = (const float*)d_in[7];

  const size_t NW0  = (size_t)8 * 1024 * 512;   // 4,194,304
  const size_t NW1  = (size_t)8 * 1024 * 1024;  // 8,388,608
  const size_t NW2  = (size_t)8 * 512 * 1024;   // 4,194,304
  const size_t NACT = (size_t)4096 * 1024;      // 4,194,304
  const size_t NX   = (size_t)4096 * 512;       // 2,097,152

  half_t* p = (half_t*)d_ws;
  half_t* wHi0 = p; p += NW0;
  half_t* wLo0 = p; p += NW0;
  half_t* wHi1 = p; p += NW1;
  half_t* wLo1 = p; p += NW1;
  half_t* wHi2 = p; p += NW2;
  half_t* wLo2 = p; p += NW2;
  half_t* aAhi = p; p += NACT;   // ping buffer (holds x, then h1-out of L1)
  half_t* aAlo = p; p += NACT;
  half_t* aBhi = p; p += NACT;   // pong buffer (holds h0)
  half_t* aBlo = p; p += NACT;
  // total ws use: ~100.7 MB

  const int T = 256;
  split_kernel<<<dim3((unsigned)(NW0 / T)), T, 0, stream>>>(W0, wHi0, wLo0, (int)NW0);
  split_kernel<<<dim3((unsigned)(NW1 / T)), T, 0, stream>>>(W1, wHi1, wLo1, (int)NW1);
  split_kernel<<<dim3((unsigned)(NW2 / T)), T, 0, stream>>>(W2, wHi2, wLo2, (int)NW2);
  split_kernel<<<dim3((unsigned)(NX  / T)), T, 0, stream>>>(x,  aAhi, aAlo, (int)NX);

  // Layer 0: K=512,  O=1024, ELU, half out   (x in aA -> h0 in aB)
  layer_kernel<512, 1024, true, false><<<dim3(8, 128), 128, 0, stream>>>(
      aAhi, aAlo, wHi0, wLo0, blend, B0, aBhi, aBlo, nullptr);
  // Layer 1: K=1024, O=1024, ELU, half out   (h0 in aB -> h1 in aA)
  layer_kernel<1024, 1024, true, false><<<dim3(8, 128), 128, 0, stream>>>(
      aBhi, aBlo, wHi1, wLo1, blend, B1, aAhi, aAlo, nullptr);
  // Layer 2: K=1024, O=512, linear, f32 out  (h1 in aA -> d_out)
  layer_kernel<1024, 512, false, true><<<dim3(4, 128), 128, 0, stream>>>(
      aAhi, aAlo, wHi2, wLo2, blend, B2, nullptr, nullptr, (float*)d_out);
}
